// PNANet_17154099380847
// MI455X (gfx1250) — matrix operand order repaired and checked by reading, not compiled, not run on hardware
//
#include <hip/hip_runtime.h>
#include <hip/hip_bf16.h>
#include <math.h>

typedef _Float16 half_t;
typedef __attribute__((ext_vector_type(16))) _Float16 v16h;
typedef __attribute__((ext_vector_type(8)))  _Float16 v8h;
typedef __attribute__((ext_vector_type(8)))  float    v8f;

#define NN 20000     // nodes
#define NE 320000    // edges
#define NG 64        // graphs
#define EMB 80

// ---------- ordered-uint encoding of float for atomic min/max ----------
__device__ __forceinline__ unsigned f2ord(float f) {
    unsigned u = __float_as_uint(f);
    return (u >> 31) ? ~u : (u | 0x80000000u);
}
__device__ __forceinline__ float ord2f(unsigned o) {
    unsigned u = (o >> 31) ? (o & 0x7FFFFFFFu) : ~o;
    return __uint_as_float(u);
}
#define ORD_PINF 0xFF800000u   // f2ord(+inf)  -> init for running min
#define ORD_NINF 0x007FFFFFu   // f2ord(-inf)  -> init for running max

// ---------- WMMA fragment helper ----------
// A (16x32 f16, M=lane&15): lo 8 halves at +0, hi 8 at +16, base kb=(lane>>4)*8
// B (32x16 f16, N=lane&15): lo 8 halves at +0, hi 8 at +16, base kb=(lane>>4)*8
__device__ __forceinline__ v8f wmma_step(const half_t* __restrict__ aptr,
                                         const half_t* __restrict__ bptr, v8f acc) {
    v8h alo = *(const v8h*)(aptr);
    v8h ahi = *(const v8h*)(aptr + 16);
    v16h a, b;
    v8h blo = *(const v8h*)(bptr);
    v8h bhi = *(const v8h*)(bptr + 16);
#pragma unroll
    for (int i = 0; i < 8; ++i) { a[i] = alo[i]; a[i + 8] = ahi[i]; b[i] = blo[i]; b[i + 8] = bhi[i]; }
    return __builtin_amdgcn_wmma_f32_16x16x32_f16(false, a, false, b, (short)0, acc,
                                                  false, false);
}

// =================== weight conversion (f32 -> transposed padded f16) ===================
__global__ void k_conv_preW(const float* __restrict__ W, half_t* __restrict__ BT,
                            int F, int Kpad) {
    int TF = 5 * F, total = TF * Kpad;
    int i = blockIdx.x * blockDim.x + threadIdx.x;
    if (i >= total) return;
    int c = i / Kpad, k = i % Kpad;
    int t = c / F, o = c % F;
    float v = (k < 3 * F) ? W[(t * 3 * F + k) * F + o] : 0.f;
    BT[i] = (half_t)v;
}
__global__ void k_conv_postW(const float* __restrict__ W, half_t* __restrict__ BT,
                             int F, int Kpad) {
    int total = 80 * Kpad;
    int i = blockIdx.x * blockDim.x + threadIdx.x;
    if (i >= total) return;
    int c = i / Kpad, k = i % Kpad;
    int t = c / 16, o = c % 16;
    float v = (k < 13 * F) ? W[(t * 13 * F + k) * 16 + o] : 0.f;
    BT[i] = (half_t)v;
}
__global__ void k_conv_linW(const float* __restrict__ W, half_t* __restrict__ BT) {
    int i = blockIdx.x * blockDim.x + threadIdx.x;
    if (i >= 80 * 96) return;
    int c = i / 96, k = i % 96;
    BT[i] = (half_t)((k < 80) ? W[k * 80 + c] : 0.f);
}
// f32 -> f16 mirror of node features (feeds async-to-LDS gathers)
__global__ void k_h16(const float* __restrict__ in, half_t* __restrict__ out, int n) {
    int i = blockIdx.x * blockDim.x + threadIdx.x;
    if (i < n) out[i] = (half_t)in[i];
}

// =================== init / degree / avg-log ===================
__global__ void k_init(float* __restrict__ sum, float* __restrict__ sq,
                       unsigned* __restrict__ mn, unsigned* __restrict__ mx,
                       float* __restrict__ cnt, int NTF, int N) {
    int i = blockIdx.x * blockDim.x + threadIdx.x;
    if (i < NTF) { sum[i] = 0.f; sq[i] = 0.f; mn[i] = ORD_PINF; mx[i] = ORD_NINF; }
    if (i < N) cnt[i] = 0.f;
}
__global__ void k_deg(const int* __restrict__ dst, float* __restrict__ cnt, int E) {
    int i = blockIdx.x * blockDim.x + threadIdx.x;
    if (i < E) atomicAdd(cnt + dst[i], 1.f);
}
__global__ void k_avglog(const float* __restrict__ cnt, float* __restrict__ out, int N) {
    __shared__ float red[256];
    float s = 0.f;
    for (int i = threadIdx.x; i < N; i += 256) s += log1pf(cnt[i]);
    red[threadIdx.x] = s;
    __syncthreads();
    for (int st = 128; st > 0; st >>= 1) {
        if (threadIdx.x < st) red[threadIdx.x] += red[threadIdx.x + st];
        __syncthreads();
    }
    if (threadIdx.x == 0) out[0] = red[0] / (float)N;
}

// =================== pre_nn: per-edge GEMM + scatter reduce ===================
// block = 256 threads (8 waves), 16 edges; A tile [16][Kpad] f16 in LDS.
// x_i|x_j gathers arrive via GLOBAL_LOAD_ASYNC_TO_LDS_B128 from the f16 feature
// mirror; edge-embedding + padding are filled by VALU. Atile is the only LDS
// object in this kernel, so its byte offsets are the wave-relative LDS addresses.
__global__ void __launch_bounds__(256)
k_pre(const half_t* __restrict__ h16, const int* __restrict__ src,
      const int* __restrict__ dst, const float* __restrict__ ea,
      const float* __restrict__ We, const float* __restrict__ be,
      const half_t* __restrict__ BT, const float* __restrict__ bpre,
      float* __restrict__ sumB, float* __restrict__ sqB,
      unsigned* __restrict__ mnB, unsigned* __restrict__ mxB,
      int E, int F, int TF, int Kpad) {
    __shared__ half_t Atile[16 * 256];          // Kpad <= 256
    int eb = blockIdx.x * 16;
    int tid = threadIdx.x;

    // --- async gather of x_i | x_j (16B granules; F*2 is a multiple of 16) ---
    int wpr = F >> 3;                   // 16-byte chunks per F-region
    int nOps = 16 * 2 * wpr;            // 256 (F=64) or 320 (F=80)
    for (int w = tid; w < nOps; w += 256) {
        int r   = w / (2 * wpr);
        int rem = w - r * 2 * wpr;
        int reg = rem / wpr;            // 0 -> x_i (dst), 1 -> x_j (src)
        int c8  = (rem - reg * wpr) * 8;
        int e = eb + r;
        if (e < E) {
            int node = reg ? src[e] : dst[e];
            unsigned ldsOff = (unsigned)((r * Kpad + reg * F + c8) * 2);
            unsigned gOff   = (unsigned)((node * F + c8) * 2);
            asm volatile("global_load_async_to_lds_b128 %0, %1, %2"
                         :: "v"(ldsOff), "v"(gOff), "s"(h16) : "memory");
        }
    }
    // --- edge embedding + zero padding via normal LDS stores ---
    int restW = Kpad - 2 * F;
    for (int i = tid; i < 16 * restW; i += 256) {
        int r = i / restW, k = 2 * F + (i - (i / restW) * restW);
        int e = eb + r;
        float v = 0.f;
        if (e < E && k < 3 * F) {
            int f = k - 2 * F;
            v = ea[e * 2] * We[f] + ea[e * 2 + 1] * We[F + f] + be[f];
        }
        Atile[r * Kpad + k] = (half_t)v;
    }
    asm volatile("s_wait_asynccnt 0x0" ::: "memory");
    __syncthreads();

    int lane = tid & 31, wave = tid >> 5;
    int row = lane & 15, kb = (lane >> 4) * 8;
    int NT = TF >> 4, KC = Kpad >> 5;
    for (int nt = wave; nt < NT; nt += 8) {
        v8f acc = {0.f, 0.f, 0.f, 0.f, 0.f, 0.f, 0.f, 0.f};
        int col = nt * 16 + row;
        const half_t* ab = Atile + row * Kpad + kb;
        const half_t* bb = BT + col * Kpad + (lane >> 4) * 8;
        __builtin_prefetch(bb, 0, 0);
        for (int kc = 0; kc < KC; ++kc)
            acc = wmma_step(ab + kc * 32, bb + kc * 32, acc);
        float bias = bpre[col];
        int mofs = (lane >> 4) * 8;
#pragma unroll
        for (int r2 = 0; r2 < 8; ++r2) {
            int e = eb + r2 + mofs;
            if (e < E) {
                float v = acc[r2] + bias;
                int o = dst[e] * TF + col;
                atomicAdd(sumB + o, v);
                atomicAdd(sqB + o, v * v);
                unsigned ov = f2ord(v);
                atomicMax(mxB + o, ov);
                atomicMin(mnB + o, ov);
            }
        }
    }
}

// =================== post_nn: per-node per-tower GEMM ===================
// block = 160 threads (5 waves = 5 towers), 16 nodes; A tile [5][16][Kpad] f16 in LDS.
__global__ void __launch_bounds__(160)
k_post(const float* __restrict__ h, const float* __restrict__ sumB,
       const float* __restrict__ sqB, const unsigned* __restrict__ mnB,
       const unsigned* __restrict__ mxB, const float* __restrict__ cnt,
       const float* __restrict__ avgLog, const half_t* __restrict__ BT,
       const float* __restrict__ bpost, float* __restrict__ postBuf,
       int N, int F, int TF, int Kpad) {
    extern __shared__ half_t dynls[];
    int nb = blockIdx.x * 16;
    int lane = threadIdx.x & 31, wave = threadIdx.x >> 5;   // wave == tower
    float al = avgLog[0];
    half_t* myT = dynls + wave * 16 * Kpad;
    int K13 = 13 * F;
    for (int i = lane; i < 16 * Kpad; i += 32) {
        int r = i / Kpad, k = i % Kpad;
        int node = nb + r;
        float v = 0.f;
        if (node < N && k < K13) {
            if (k < F) v = h[node * F + k];
            else {
                int s = k - F;
                int grp = s / (4 * F);          // 0 identity, 1 amplify, 2 attenuate
                int w = s - grp * 4 * F;
                int kind = w / F;               // 0 mean, 1 min, 2 max, 3 std
                int f = w - kind * F;
                int idx = node * TF + wave * F + f;
                float c_ = cnt[node];
                float den = fmaxf(c_, 1.f);
                float agg;
                if (kind == 0) agg = sumB[idx] / den;
                else if (kind == 1) agg = (c_ > 0.f) ? ord2f(mnB[idx]) : 0.f;
                else if (kind == 2) agg = (c_ > 0.f) ? ord2f(mxB[idx]) : 0.f;
                else {
                    float mean = sumB[idx] / den, m2 = sqB[idx] / den;
                    agg = sqrtf(fmaxf(m2 - mean * mean, 0.f) + 1e-5f);
                }
                float amp = log1pf(fmaxf(c_, 1.f)) / al;
                v = (grp == 0) ? agg : (grp == 1 ? agg * amp : agg / amp);
            }
        }
        myT[i] = (half_t)v;
    }
    __syncthreads();
    int row = lane & 15, kb = (lane >> 4) * 8;
    const half_t* ab = myT + row * Kpad + kb;
    const half_t* bb = BT + (wave * 16 + row) * Kpad + (lane >> 4) * 8;
    __builtin_prefetch(bb, 0, 0);
    v8f acc = {0.f, 0.f, 0.f, 0.f, 0.f, 0.f, 0.f, 0.f};
    int KC = Kpad >> 5;
    for (int kc = 0; kc < KC; ++kc)
        acc = wmma_step(ab + kc * 32, bb + kc * 32, acc);
    float bias = bpost[wave * 16 + row];
#pragma unroll
    for (int r2 = 0; r2 < 8; ++r2) {
        int node = nb + r2 + kb;
        if (node < N) postBuf[node * EMB + wave * 16 + row] = acc[r2] + bias;
    }
}

// =================== Wlin: [N,80]@[80,80] + bias (+relu), dual f32/f16 output ===================
__global__ void __launch_bounds__(160)
k_lin(const float* __restrict__ inBuf, const half_t* __restrict__ BT,
      const float* __restrict__ blin, float* __restrict__ outBuf,
      half_t* __restrict__ out16, int N, int relu) {
    __shared__ half_t Atile[16 * 96];
    int nb = blockIdx.x * 16;
    int tid = threadIdx.x;
    for (int i = tid; i < 16 * 96; i += 160) {
        int r = i / 96, k = i % 96;
        int node = nb + r;
        Atile[i] = (half_t)((node < N && k < 80) ? inBuf[node * EMB + k] : 0.f);
    }
    __syncthreads();
    int lane = tid & 31, wave = tid >> 5;       // wave = N-tile (5 tiles of 16)
    int row = lane & 15, kb = (lane >> 4) * 8;
    int col = wave * 16 + row;
    const half_t* ab = Atile + row * 96 + kb;
    const half_t* bb = BT + col * 96 + (lane >> 4) * 8;
    v8f acc = {0.f, 0.f, 0.f, 0.f, 0.f, 0.f, 0.f, 0.f};
    for (int kc = 0; kc < 3; ++kc)
        acc = wmma_step(ab + kc * 32, bb + kc * 32, acc);
    float bias = blin[col];
#pragma unroll
    for (int r2 = 0; r2 < 8; ++r2) {
        int node = nb + r2 + kb;
        if (node < N) {
            float v = acc[r2] + bias;
            if (relu) v = fmaxf(v, 0.f);
            outBuf[node * EMB + col] = v;
            out16[node * EMB + col] = (half_t)v;
        }
    }
}

// =================== pooling / MLP / loss ===================
__global__ void k_zero(float* __restrict__ p, int n) {
    int i = blockIdx.x * blockDim.x + threadIdx.x;
    if (i < n) p[i] = 0.f;
}
__global__ void k_pool(const float* __restrict__ h, const int* __restrict__ batch,
                       float* __restrict__ hg, int N) {
    int i = blockIdx.x * blockDim.x + threadIdx.x;
    if (i < N * EMB) {
        int node = i / EMB, c = i % EMB;
        atomicAdd(hg + batch[node] * EMB + c, h[i]);
    }
}
__global__ void k_dense(const float* __restrict__ in, const float* __restrict__ W,
                        const float* __restrict__ b, float* __restrict__ out,
                        int G, int K, int No, int relu) {
    int i = blockIdx.x * blockDim.x + threadIdx.x;
    if (i >= G * No) return;
    int g = i / No, o = i % No;
    float s = b[o];
    for (int k = 0; k < K; ++k) s += in[g * K + k] * W[k * No + o];
    if (relu) s = fmaxf(s, 0.f);
    out[i] = s;
}
__global__ void k_loss(const float* __restrict__ pred, const float* __restrict__ y,
                       float* __restrict__ out) {
    __shared__ float red[64];
    int t = threadIdx.x;
    float d = pred[t] - y[t];
    red[t] = d * d;
    __syncthreads();
    for (int st = 32; st > 0; st >>= 1) {
        if (t < st) red[t] += red[t + st];
        __syncthreads();
    }
    if (t == 0) out[0] = red[0] / 64.f;
}

// =================== host ===================
extern "C" void kernel_launch(void* const* d_in, const int* in_sizes, int n_in,
                              void* d_out, int out_size, void* d_ws, size_t ws_size,
                              hipStream_t stream) {
    (void)in_sizes; (void)n_in; (void)out_size; (void)ws_size;
    const float* x     = (const float*)d_in[0];
    const int*   eidx  = (const int*)d_in[1];
    const int*   srcI  = eidx;
    const int*   dstI  = eidx + NE;
    const float* ea    = (const float*)d_in[2];
    const int*   batch = (const int*)d_in[3];
    const float* y     = (const float*)d_in[4];
    const float* Wf = (const float*)d_in[21];
    const float* bf = (const float*)d_in[22];

    // workspace carve-out (deterministic; needs ~148 MB)
    size_t off = 0;
    char* base = (char*)d_ws;
    auto alloc = [&](size_t bytes) -> void* {
        void* p = base + off;
        off += (bytes + 255) & ~(size_t)255;
        return p;
    };
    float*    sumB    = (float*)alloc((size_t)NN * 400 * 4);
    float*    sqB     = (float*)alloc((size_t)NN * 400 * 4);
    unsigned* mnB     = (unsigned*)alloc((size_t)NN * 400 * 4);
    unsigned* mxB     = (unsigned*)alloc((size_t)NN * 400 * 4);
    float*    cnt     = (float*)alloc(NN * 4);
    float*    avgL    = (float*)alloc(256);
    float*    postBuf = (float*)alloc((size_t)NN * EMB * 4);
    float*    hbuf    = (float*)alloc((size_t)NN * EMB * 4);
    half_t*   x16     = (half_t*)alloc((size_t)NN * 64 * 2);
    half_t*   h16     = (half_t*)alloc((size_t)NN * EMB * 2);
    half_t*   BpreT   = (half_t*)alloc(400 * 256 * 2);
    half_t*   BpostT  = (half_t*)alloc(80 * 1056 * 2);
    half_t*   BlinT   = (half_t*)alloc(80 * 96 * 2);
    float*    zA      = (float*)alloc(64 * 512 * 4);
    float*    zB      = (float*)alloc(64 * 512 * 4);

    k_h16<<<(NN * 64 + 255) / 256, 256, 0, stream>>>(x, x16, NN * 64);

    for (int l = 0; l < 2; ++l) {
        int F = l ? 80 : 64;
        int TF = 5 * F;
        int KpreP  = l ? 256 : 192;    // pad(3F, 32)
        int KpostP = l ? 1056 : 832;   // pad(13F, 32)
        int pb = 5 + 8 * l;
        const float* We    = (const float*)d_in[pb + 0];
        const float* be    = (const float*)d_in[pb + 1];
        const float* Wpre  = (const float*)d_in[pb + 2];
        const float* bpre  = (const float*)d_in[pb + 3];
        const float* Wpost = (const float*)d_in[pb + 4];
        const float* bpost = (const float*)d_in[pb + 5];
        const float* Wlin  = (const float*)d_in[pb + 6];
        const float* blin  = (const float*)d_in[pb + 7];
        const float*  hin   = l ? hbuf : x;
        const half_t* hin16 = l ? h16 : x16;

        k_conv_preW<<<(TF * KpreP + 255) / 256, 256, 0, stream>>>(Wpre, BpreT, F, KpreP);
        k_conv_postW<<<(80 * KpostP + 255) / 256, 256, 0, stream>>>(Wpost, BpostT, F, KpostP);
        k_conv_linW<<<30, 256, 0, stream>>>(Wlin, BlinT);
        k_init<<<(NN * TF + 255) / 256, 256, 0, stream>>>(sumB, sqB, mnB, mxB, cnt, NN * TF, NN);
        k_deg<<<(NE + 255) / 256, 256, 0, stream>>>(dstI, cnt, NE);
        k_avglog<<<1, 256, 0, stream>>>(cnt, avgL, NN);
        k_pre<<<NE / 16, 256, 0, stream>>>(hin16, srcI, dstI, ea, We, be, BpreT, bpre,
                                           sumB, sqB, mnB, mxB, NE, F, TF, KpreP);
        k_post<<<NN / 16, 160, (size_t)5 * 16 * KpostP * 2, stream>>>(
            hin, sumB, sqB, mnB, mxB, cnt, avgL, BpostT, bpost, postBuf, NN, F, TF, KpostP);
        k_lin<<<NN / 16, 160, 0, stream>>>(postBuf, BlinT, blin, hbuf, h16, NN,
                                           l == 0 ? 1 : 0);
    }

    // global_add_pool + first_MLP + MLP + loss
    float* out = (float*)d_out;
    k_zero<<<(NG * EMB + 255) / 256, 256, 0, stream>>>(zA, NG * EMB);
    k_pool<<<(NN * EMB + 255) / 256, 256, 0, stream>>>(hbuf, batch, zA, NN);
    k_dense<<<(NG * 512 + 255) / 256, 256, 0, stream>>>(zA, Wf, bf, zB, NG, 80, 512, 0);
    k_dense<<<(NG * 256 + 255) / 256, 256, 0, stream>>>(
        zB, (const float*)d_in[23], (const float*)d_in[24], zA, NG, 512, 256, 1);
    k_dense<<<(NG * 128 + 255) / 256, 256, 0, stream>>>(
        zA, (const float*)d_in[25], (const float*)d_in[26], zB, NG, 256, 128, 1);
    k_dense<<<(NG * 64 + 255) / 256, 256, 0, stream>>>(
        zB, (const float*)d_in[27], (const float*)d_in[28], zA, NG, 128, 64, 1);
    k_dense<<<1, 256, 0, stream>>>(
        zA, (const float*)d_in[29], (const float*)d_in[30], out, NG, 64, 1, 0);
    k_loss<<<1, 64, 0, stream>>>(out, y, out + 64);
}
